// Qwen3MoeDecoderLayer_4260607558197
// MI455X (gfx1250) — compile-verified, matrix operand order checked by reading, not run on hardware
//
#include <hip/hip_runtime.h>
#include <math.h>

// ---------------- problem constants ----------------
static constexpr int S_   = 1024;
static constexpr int H_   = 2048;
static constexpr int NH_  = 16;
static constexpr int NKV_ = 4;
static constexpr int HD_  = 128;
static constexpr int E_   = 8;
static constexpr int F_   = 768;
static constexpr float EPS_ = 1e-6f;

// ---------------- WMMA types ----------------
typedef __bf16 bf16x16 __attribute__((ext_vector_type(16)));
typedef float  f32x8   __attribute__((ext_vector_type(8)));

union FragB { bf16x16 v; unsigned u[8]; };

__device__ __forceinline__ f32x8 wmma_bf16(const FragB& a, const FragB& b, f32x8 c) {
    return __builtin_amdgcn_wmma_f32_16x16x32_bf16(false, a.v, false, b.v,
                                                   (short)0, c, false, false);
}

// K-offset inside a 16x32 bf16 A/B fragment for VGPR v, lane-half h.
__device__ __forceinline__ int koff(int v, int half) {
    return ((v & 3) * 2) + half * 8 + ((v >> 2) * 16);
}

// fp32 pair -> packed bf16 pair: single native CDNA5 instruction (RNE).
__device__ __forceinline__ unsigned pack2bf(float a, float b) {
    unsigned r;
    asm("v_cvt_pk_bf16_f32 %0, %1, %2" : "=v"(r) : "v"(a), "v"(b));
    return r;  // lo = bf16(a), hi = bf16(b)
}
__device__ __forceinline__ unsigned short f2bf(float a) {
    return (unsigned short)((__float_as_uint(a) + 0x8000u) >> 16);
}
__device__ __forceinline__ float bf2f(unsigned short h) {
    return __uint_as_float(((unsigned)h) << 16);
}

// ---------------- RMSNorm (fp32 in -> bf16 out) ----------------
__global__ __launch_bounds__(256) void rmsnorm_bf16(
    const float* __restrict__ x, const float* __restrict__ w,
    unsigned short* __restrict__ y)
{
    int row = blockIdx.x;
    const float* xr = x + (size_t)row * H_;
    __shared__ float red[8];
    float s = 0.f;
    for (int i = threadIdx.x; i < H_; i += 256) { float v = xr[i]; s += v * v; }
    for (int off = 16; off > 0; off >>= 1) s += __shfl_xor(s, off, 32);
    if ((threadIdx.x & 31) == 0) red[threadIdx.x >> 5] = s;
    __syncthreads();
    float tot = 0.f;
    #pragma unroll
    for (int i = 0; i < 8; i++) tot += red[i];
    float r = rsqrtf(tot / (float)H_ + EPS_);
    for (int i = threadIdx.x; i < H_; i += 256)
        y[(size_t)row * H_ + i] = f2bf(xr[i] * r * w[i]);
}

// ---------------- generic GEMM: C[M,N] = A_bf16[M,K] * W_f32[N,K]^T (+addend) ----
// block = 256 thr = 8 waves; wave tile 16x64; block tile 64(M) x 128(N)
__global__ __launch_bounds__(256) void gemm_bf16(
    const unsigned short* __restrict__ A, const float* __restrict__ W,
    float* __restrict__ C, const float* __restrict__ addend,
    int M, int N, int K)
{
    int lane = threadIdx.x & 31, wave = threadIdx.x >> 5;
    int wm = wave & 3, wn = wave >> 2;
    int row0 = blockIdx.y * 64 + wm * 16;
    int col0 = blockIdx.x * 128 + wn * 64;
    int half = lane >> 4, lm = lane & 15;

    f32x8 acc[4];
    #pragma unroll
    for (int j = 0; j < 4; j++)
        #pragma unroll
        for (int r = 0; r < 8; r++) acc[j][r] = 0.f;

    for (int k0 = 0; k0 < K; k0 += 32) {
        FragB a;
        const unsigned short* Ar = A + (size_t)(row0 + lm) * K + k0;
        #pragma unroll
        for (int v = 0; v < 8; v++) a.u[v] = *(const unsigned*)(Ar + koff(v, half));
        #pragma unroll
        for (int j = 0; j < 4; j++) {
            int n = col0 + j * 16 + lm;
            const float* Wr = W + (size_t)n * K + k0;
            FragB b;
            #pragma unroll
            for (int v = 0; v < 8; v++) { int k = koff(v, half); b.u[v] = pack2bf(Wr[k], Wr[k + 1]); }
            acc[j] = wmma_bf16(a, b, acc[j]);
        }
    }
    #pragma unroll
    for (int j = 0; j < 4; j++)
        #pragma unroll
        for (int r = 0; r < 8; r++) {
            int row = row0 + half * 8 + r;
            int col = col0 + j * 16 + lm;
            float v = acc[j][r];
            if (addend) v += addend[(size_t)row * N + col];
            C[(size_t)row * N + col] = v;
        }
}

// ---------------- per-head RMSNorm + RoPE -> bf16 [NHX,S,HD] ----------------
__global__ __launch_bounds__(128) void rope_norm(
    const float* __restrict__ in,   // [S, NHX*HD]
    const float* __restrict__ nw,   // [HD]
    const float* __restrict__ cosb, const float* __restrict__ sinb, // [S,HD]
    unsigned short* __restrict__ out, // [NHX,S,HD]
    int NHX)
{
    int s = blockIdx.x, h = blockIdx.y, d = threadIdx.x;
    __shared__ float yv[HD_];
    __shared__ float red[4];
    float x = in[(size_t)s * (NHX * HD_) + h * HD_ + d];
    float sq = x * x;
    for (int off = 16; off > 0; off >>= 1) sq += __shfl_xor(sq, off, 32);
    if ((d & 31) == 0) red[d >> 5] = sq;
    __syncthreads();
    float tot = red[0] + red[1] + red[2] + red[3];
    float r = rsqrtf(tot / (float)HD_ + EPS_);
    float y = x * r * nw[d];
    yv[d] = y;
    __syncthreads();
    float rot = (d < HD_ / 2) ? -yv[d + HD_ / 2] : yv[d - HD_ / 2];
    float o = y * cosb[(size_t)s * HD_ + d] + rot * sinb[(size_t)s * HD_ + d];
    out[((size_t)h * S_ + s) * HD_ + d] = f2bf(o);
}

// ---------------- V pack: fp32 [S,NKV*HD] -> bf16 transposed [NKV,HD,S] -------
__global__ __launch_bounds__(256) void v_pack(
    const float* __restrict__ v, unsigned short* __restrict__ vT)
{
    int idx = blockIdx.x * 256 + threadIdx.x;
    if (idx >= NKV_ * HD_ * S_) return;
    int s = idx % S_;
    int d = (idx / S_) % HD_;
    int kh = idx / (S_ * HD_);
    vT[idx] = f2bf(v[(size_t)s * (NKV_ * HD_) + kh * HD_ + d]);
}

// ---------------- flash attention, 1 wave per (head, 16-query tile) ----------
__global__ __launch_bounds__(32) void attn(
    const unsigned short* __restrict__ qb,  // [NH,S,HD]
    const unsigned short* __restrict__ kb,  // [NKV,S,HD]
    const unsigned short* __restrict__ vT,  // [NKV,HD,S]
    unsigned short* __restrict__ ob)        // [S, NH*HD] bf16
{
    const int h = blockIdx.y;
    const int kvh = h / (NH_ / NKV_);
    const int q0 = blockIdx.x * 16;
    const int lane = threadIdx.x, half = lane >> 4, lm = lane & 15;
    const float scale = 0.08838834764831845f; // 1/sqrt(128)

    __shared__ unsigned short P[16 * 32];

    FragB qf[4];
    const unsigned short* qrow = qb + ((size_t)h * S_ + (q0 + lm)) * HD_;
    #pragma unroll
    for (int c = 0; c < 4; c++)
        #pragma unroll
        for (int v = 0; v < 8; v++) qf[c].u[v] = *(const unsigned*)(qrow + c * 32 + koff(v, half));

    f32x8 oacc[8];
    #pragma unroll
    for (int dc = 0; dc < 8; dc++)
        #pragma unroll
        for (int r = 0; r < 8; r++) oacc[dc][r] = 0.f;
    float rowmax[8], rowsum[8];
    #pragma unroll
    for (int r = 0; r < 8; r++) { rowmax[r] = -INFINITY; rowsum[r] = 0.f; }

    for (int kb0 = 0; kb0 < q0 + 16; kb0 += 32) {
        f32x8 s0, s1;
        #pragma unroll
        for (int r = 0; r < 8; r++) { s0[r] = 0.f; s1[r] = 0.f; }
        #pragma unroll
        for (int c = 0; c < 4; c++) {
            FragB b0, b1;
            const unsigned short* kr0 = kb + ((size_t)kvh * S_ + (kb0 + lm)) * HD_ + c * 32;
            const unsigned short* kr1 = kb + ((size_t)kvh * S_ + (kb0 + 16 + lm)) * HD_ + c * 32;
            #pragma unroll
            for (int v = 0; v < 8; v++) {
                int k = koff(v, half);
                b0.u[v] = *(const unsigned*)(kr0 + k);
                b1.u[v] = *(const unsigned*)(kr1 + k);
            }
            s0 = wmma_bf16(qf[c], b0, s0);
            s1 = wmma_bf16(qf[c], b1, s1);
        }
        // online softmax per row (row = half*8 + r, key cols across 16-lane halves)
        #pragma unroll
        for (int r = 0; r < 8; r++) {
            int qi = q0 + half * 8 + r;
            float v0 = (kb0 + lm      <= qi) ? s0[r] * scale : -1e30f;
            float v1 = (kb0 + 16 + lm <= qi) ? s1[r] * scale : -1e30f;
            float mx = fmaxf(v0, v1);
            for (int off = 1; off < 16; off <<= 1) mx = fmaxf(mx, __shfl_xor(mx, off, 16));
            float nm = fmaxf(rowmax[r], mx);
            float p0 = __expf(v0 - nm), p1 = __expf(v1 - nm);
            float rs = __expf(rowmax[r] - nm);
            float ps = p0 + p1;
            for (int off = 1; off < 16; off <<= 1) ps += __shfl_xor(ps, off, 16);
            rowsum[r] = rowsum[r] * rs + ps;
            rowmax[r] = nm;
            #pragma unroll
            for (int dc = 0; dc < 8; dc++) oacc[dc][r] *= rs;
            P[(half * 8 + r) * 32 + lm]      = f2bf(p0);
            P[(half * 8 + r) * 32 + 16 + lm] = f2bf(p1);
        }
        __syncthreads();
        FragB pf;  // A-layout read of the 16x32 P tile
        #pragma unroll
        for (int v = 0; v < 8; v++) pf.u[v] = *(const unsigned*)&P[lm * 32 + koff(v, half)];
        __syncthreads();
        #pragma unroll
        for (int dc = 0; dc < 8; dc++) {
            FragB bv;
            const unsigned short* vr = vT + ((size_t)kvh * HD_ + dc * 16 + lm) * S_ + kb0;
            #pragma unroll
            for (int v = 0; v < 8; v++) bv.u[v] = *(const unsigned*)(vr + koff(v, half));
            oacc[dc] = wmma_bf16(pf, bv, oacc[dc]);
        }
    }
    #pragma unroll
    for (int dc = 0; dc < 8; dc++)
        #pragma unroll
        for (int r = 0; r < 8; r++) {
            int qi = q0 + half * 8 + r;
            ob[(size_t)qi * (NH_ * HD_) + h * HD_ + dc * 16 + lm] = f2bf(oacc[dc][r] / rowsum[r]);
        }
}

// ---------------- router: logits, softmax, top-2, combine weights ------------
__global__ __launch_bounds__(32) void gate_topk(
    const unsigned short* __restrict__ h2, const float* __restrict__ gw,
    float* __restrict__ combine)
{
    int t = blockIdx.x, lane = threadIdx.x;
    float lg[E_];
    const unsigned short* hr = h2 + (size_t)t * H_;
    #pragma unroll
    for (int e = 0; e < E_; e++) {
        const float* wr = gw + (size_t)e * H_;
        float s = 0.f;
        for (int i = lane; i < H_; i += 32) s += bf2f(hr[i]) * wr[i];
        for (int off = 16; off > 0; off >>= 1) s += __shfl_xor(s, off, 32);
        lg[e] = s;
    }
    float mx = lg[0];
    #pragma unroll
    for (int e = 1; e < E_; e++) mx = fmaxf(mx, lg[e]);
    float p[E_]; float sum = 0.f;
    #pragma unroll
    for (int e = 0; e < E_; e++) { p[e] = __expf(lg[e] - mx); sum += p[e]; }
    #pragma unroll
    for (int e = 0; e < E_; e++) p[e] /= sum;
    int i1 = 0;
    #pragma unroll
    for (int e = 1; e < E_; e++) if (p[e] > p[i1]) i1 = e;
    int i2 = (i1 == 0) ? 1 : 0;
    #pragma unroll
    for (int e = 0; e < E_; e++) if (e != i1 && p[e] > p[i2]) i2 = e;
    float norm = p[i1] + p[i2];
    if (lane < E_)
        combine[(size_t)t * E_ + lane] =
            (lane == i1) ? p[i1] / norm : ((lane == i2) ? p[i2] / norm : 0.f);
}

// ---------------- fused gate/up GEMM + SiLU -> bf16 act [E,S,F] --------------
__global__ __launch_bounds__(256) void moe_gu(
    const unsigned short* __restrict__ h2,
    const float* __restrict__ gate_proj, const float* __restrict__ up_proj,
    unsigned short* __restrict__ act)
{
    int e = blockIdx.z;
    int lane = threadIdx.x & 31, wave = threadIdx.x >> 5;
    int row0 = blockIdx.y * 128 + wave * 16;
    int col0 = blockIdx.x * 64;
    int half = lane >> 4, lm = lane & 15;
    const float* gp = gate_proj + (size_t)e * F_ * H_;
    const float* up = up_proj   + (size_t)e * F_ * H_;

    f32x8 ag[4], au[4];
    #pragma unroll
    for (int j = 0; j < 4; j++)
        #pragma unroll
        for (int r = 0; r < 8; r++) { ag[j][r] = 0.f; au[j][r] = 0.f; }

    for (int k0 = 0; k0 < H_; k0 += 32) {
        FragB a;
        const unsigned short* Ar = h2 + (size_t)(row0 + lm) * H_ + k0;
        #pragma unroll
        for (int v = 0; v < 8; v++) a.u[v] = *(const unsigned*)(Ar + koff(v, half));
        #pragma unroll
        for (int j = 0; j < 4; j++) {
            int n = col0 + j * 16 + lm;
            const float* g1 = gp + (size_t)n * H_ + k0;
            const float* u1 = up + (size_t)n * H_ + k0;
            FragB bg, bu;
            #pragma unroll
            for (int v = 0; v < 8; v++) {
                int k = koff(v, half);
                bg.u[v] = pack2bf(g1[k], g1[k + 1]);
                bu.u[v] = pack2bf(u1[k], u1[k + 1]);
            }
            ag[j] = wmma_bf16(a, bg, ag[j]);
            au[j] = wmma_bf16(a, bu, au[j]);
        }
    }
    #pragma unroll
    for (int j = 0; j < 4; j++)
        #pragma unroll
        for (int r = 0; r < 8; r++) {
            float g = ag[j][r], u = au[j][r];
            float sg = g / (1.f + __expf(-g));
            int row = row0 + half * 8 + r, col = col0 + j * 16 + lm;
            act[((size_t)e * S_ + row) * F_ + col] = f2bf(sg * u);
        }
}

// ---------------- down-proj over all experts + combine + residual ------------
__global__ __launch_bounds__(256) void moe_down(
    const unsigned short* __restrict__ act,   // [E,S,F] bf16
    const float* __restrict__ down_proj,      // [E,H,F]
    const float* __restrict__ combine,        // [S,E]
    const float* __restrict__ x,              // [S,H]
    float* __restrict__ out)                  // [S,H]
{
    int lane = threadIdx.x & 31, wave = threadIdx.x >> 5;
    int row0 = blockIdx.y * 128 + wave * 16;
    int col0 = blockIdx.x * 64;
    int half = lane >> 4, lm = lane & 15;

    f32x8 acc[4];
    #pragma unroll
    for (int j = 0; j < 4; j++)
        #pragma unroll
        for (int r = 0; r < 8; r++)
            acc[j][r] = x[(size_t)(row0 + half * 8 + r) * H_ + col0 + j * 16 + lm];

    for (int e = 0; e < E_; e++) {
        f32x8 ye[4];
        #pragma unroll
        for (int j = 0; j < 4; j++)
            #pragma unroll
            for (int r = 0; r < 8; r++) ye[j][r] = 0.f;
        const unsigned short* A = act + (size_t)e * S_ * F_;
        const float* W = down_proj + (size_t)e * H_ * F_;
        for (int k0 = 0; k0 < F_; k0 += 32) {
            FragB a;
            const unsigned short* Ar = A + (size_t)(row0 + lm) * F_ + k0;
            #pragma unroll
            for (int v = 0; v < 8; v++) a.u[v] = *(const unsigned*)(Ar + koff(v, half));
            #pragma unroll
            for (int j = 0; j < 4; j++) {
                int n = col0 + j * 16 + lm;
                const float* Wr = W + (size_t)n * F_ + k0;
                FragB b;
                #pragma unroll
                for (int v = 0; v < 8; v++) { int k = koff(v, half); b.u[v] = pack2bf(Wr[k], Wr[k + 1]); }
                ye[j] = wmma_bf16(a, b, ye[j]);
            }
        }
        #pragma unroll
        for (int r = 0; r < 8; r++) {
            float w = combine[(size_t)(row0 + half * 8 + r) * E_ + e];
            #pragma unroll
            for (int j = 0; j < 4; j++) acc[j][r] += w * ye[j][r];
        }
    }
    #pragma unroll
    for (int j = 0; j < 4; j++)
        #pragma unroll
        for (int r = 0; r < 8; r++)
            out[(size_t)(row0 + half * 8 + r) * H_ + col0 + j * 16 + lm] = acc[j][r];
}

// ---------------- host launch ----------------
extern "C" void kernel_launch(void* const* d_in, const int* in_sizes, int n_in,
                              void* d_out, int out_size, void* d_ws, size_t ws_size,
                              hipStream_t stream) {
    const float* hidden   = (const float*)d_in[0];
    const float* cosb     = (const float*)d_in[1];
    const float* sinb     = (const float*)d_in[2];
    const float* ln1_w    = (const float*)d_in[3];
    const float* q_w      = (const float*)d_in[4];
    const float* k_w      = (const float*)d_in[5];
    const float* v_w      = (const float*)d_in[6];
    const float* o_w      = (const float*)d_in[7];
    const float* qn_w     = (const float*)d_in[8];
    const float* kn_w     = (const float*)d_in[9];
    const float* ln2_w    = (const float*)d_in[10];
    const float* gate_w   = (const float*)d_in[11];
    const float* gate_proj= (const float*)d_in[12];
    const float* up_proj  = (const float*)d_in[13];
    const float* down_proj= (const float*)d_in[14];
    float* out = (float*)d_out;

    // workspace layout
    char* p = (char*)d_ws;
    auto take = [&](size_t bytes) { char* r = p; p += (bytes + 255) & ~size_t(255); return r; };
    unsigned short* hb   = (unsigned short*)take((size_t)S_ * H_ * 2);            // bf16 ln1
    float*          qtmp = (float*)take((size_t)S_ * NH_ * HD_ * 4);
    float*          ktmp = (float*)take((size_t)S_ * NKV_ * HD_ * 4);
    float*          vtmp = (float*)take((size_t)S_ * NKV_ * HD_ * 4);
    unsigned short* qb   = (unsigned short*)take((size_t)NH_ * S_ * HD_ * 2);     // [NH,S,HD]
    unsigned short* kbuf = (unsigned short*)take((size_t)NKV_ * S_ * HD_ * 2);    // [NKV,S,HD]
    unsigned short* vT   = (unsigned short*)take((size_t)NKV_ * HD_ * S_ * 2);    // [NKV,HD,S]
    unsigned short* ob   = (unsigned short*)take((size_t)S_ * NH_ * HD_ * 2);     // attn out bf16
    float*          xres = (float*)take((size_t)S_ * H_ * 4);                     // x = resid + o@oW^T
    unsigned short* h2   = (unsigned short*)take((size_t)S_ * H_ * 2);            // bf16 ln2
    float*          comb = (float*)take((size_t)S_ * E_ * 4);
    unsigned short* act  = (unsigned short*)take((size_t)E_ * S_ * F_ * 2);

    // 1) ln1
    rmsnorm_bf16<<<S_, 256, 0, stream>>>(hidden, ln1_w, hb);
    // 2) q/k/v projections
    gemm_bf16<<<dim3((NH_*HD_)/128, S_/64), 256, 0, stream>>>(hb, q_w, qtmp, nullptr, S_, NH_*HD_, H_);
    gemm_bf16<<<dim3((NKV_*HD_)/128, S_/64), 256, 0, stream>>>(hb, k_w, ktmp, nullptr, S_, NKV_*HD_, H_);
    gemm_bf16<<<dim3((NKV_*HD_)/128, S_/64), 256, 0, stream>>>(hb, v_w, vtmp, nullptr, S_, NKV_*HD_, H_);
    // 3) per-head rmsnorm + rope, pack V transposed
    rope_norm<<<dim3(S_, NH_), 128, 0, stream>>>(qtmp, qn_w, cosb, sinb, qb, NH_);
    rope_norm<<<dim3(S_, NKV_), 128, 0, stream>>>(ktmp, kn_w, cosb, sinb, kbuf, NKV_);
    v_pack<<<(NKV_*HD_*S_ + 255)/256, 256, 0, stream>>>(vtmp, vT);
    // 4) causal flash attention
    attn<<<dim3(S_/16, NH_), 32, 0, stream>>>(qb, kbuf, vT, ob);
    // 5) o-projection + residual
    gemm_bf16<<<dim3(H_/128, S_/64), 256, 0, stream>>>(ob, o_w, xres, hidden, S_, H_, NH_*HD_);
    // 6) ln2
    rmsnorm_bf16<<<S_, 256, 0, stream>>>(xres, ln2_w, h2);
    // 7) router
    gate_topk<<<S_, 32, 0, stream>>>(h2, gate_w, comb);
    // 8) fused gate/up + SiLU
    moe_gu<<<dim3(F_/64, S_/128, E_), 256, 0, stream>>>(h2, gate_proj, up_proj, act);
    // 9) down-proj + combine + residual -> output
    moe_down<<<dim3(H_/64, S_/128), 256, 0, stream>>>(act, down_proj, comb, xres, out);
}
